// ChannelSpatialAttention2_26130581029572
// MI455X (gfx1250) — matrix operand order repaired and checked by reading, not compiled, not run on hardware
//
#include <hip/hip_runtime.h>
#include <hip/hip_bf16.h>
#include <math.h>

typedef __attribute__((ext_vector_type(16))) __bf16 v16bf;
typedef __attribute__((ext_vector_type(8)))  float  v8f;

#define NB   16
#define CCH  128
#define HWPX 16384          // 128*128
#define CHW  (CCH*HWPX)
#define EPSV 1e-5f

// workspace offsets (in floats)
#define WS_POOLED 0         // 16*256  channel-pool sums
#define WS_WCH    4096      // 16*128
#define WS_AVGS   6144      // 16*128  sum of g per (n,c)
#define WS_GMAX   8192      // 16*128  max of g per (n,c) (float bits, >=0)
#define WS_CA     10240     // 16*128
#define WS_SPMEAN 12288     // 16*16384
#define WS_SPMAX  274432    // 16*16384
#define WS_SA     536576    // 16*16384

__device__ __forceinline__ float sigmoidf(float x) { return 1.0f / (1.0f + expf(-x)); }

// ---------------- init: zero the atomic accumulators ----------------
__global__ void k_init(float* ws) {
    int i = blockIdx.x * 256 + threadIdx.x;
    if (i < 4096) ws[WS_AVGS + i] = 0.0f;   // covers WS_AVGS + WS_GMAX
}

// ---------------- global average pool of [f_vi; f_ir] ----------------
__global__ void k_pool(const float* __restrict__ f_vi, const float* __restrict__ f_ir,
                       float* ws) {
    int n = blockIdx.x >> 8;
    int c = blockIdx.x & 255;
    const float* src = (c < CCH) ? (f_vi + (size_t)n * CHW + (size_t)c * HWPX)
                                 : (f_ir + (size_t)n * CHW + (size_t)(c - CCH) * HWPX);
    float s = 0.f;
    for (int i = threadIdx.x; i < HWPX; i += 256) s += src[i];
    __shared__ float red[256];
    red[threadIdx.x] = s;
    __syncthreads();
    for (int off = 128; off > 0; off >>= 1) {
        if (threadIdx.x < off) red[threadIdx.x] += red[threadIdx.x + off];
        __syncthreads();
    }
    if (threadIdx.x == 0) ws[WS_POOLED + blockIdx.x] = red[0];
}

// ---------------- channel attention (2 tiny FCs + BN) -> wch ----------------
__global__ void k_attn1(float* ws,
                        const float* ca1_w, const float* ca1_b,
                        const float* ag, const float* ab, const float* am, const float* av,
                        const float* ca2_w, const float* ca2_b,
                        const float* bg, const float* bb, const float* bm, const float* bv) {
    __shared__ float h[16 * 64];
    int t = threadIdx.x;
    for (int idx = t; idx < 16 * 64; idx += 256) {
        int n = idx >> 6, j = idx & 63;
        float acc = ca1_b[j];
        const float* pm = ws + WS_POOLED + n * 256;
        const float* wr = ca1_w + j * 256;
        for (int c = 0; c < 256; ++c) acc += (pm[c] * (1.f / HWPX)) * wr[c];
        acc = (acc - am[j]) * (ag[j] * rsqrtf(av[j] + EPSV)) + ab[j];
        h[idx] = fmaxf(acc, 0.f);
    }
    __syncthreads();
    for (int idx = t; idx < 16 * 128; idx += 256) {
        int n = idx >> 7, o = idx & 127;
        float acc = ca2_b[o];
        const float* wr = ca2_w + o * 64;
        for (int j = 0; j < 64; ++j) acc += h[n * 64 + j] * wr[j];
        acc = (acc - bm[o]) * (bg[o] * rsqrtf(bv[o] + EPSV)) + bb[o];
        ws[WS_WCH + idx] = sigmoidf(acc);
    }
}

// ---------------- fused conv1x1 GEMM (bf16 WMMA) + BN/ReLU + all g-reductions ----------------
// grid: 16 batches * 32 chunks; block: 256 threads (8 waves); each block: 512 pixels (4 tiles of 128)
__global__ void __launch_bounds__(256)
k_main(const float* __restrict__ f_vi, const float* __restrict__ f_ir,
       const float* __restrict__ conv1_w, const float* __restrict__ conv1_b,
       const float* cg, const float* cb, const float* cm, const float* cv,
       float* ws) {
    __shared__ __bf16 wlds[CCH * 256];   // 64 KB : W' = conv1_w * bn_scale, [o][c]
    __shared__ __bf16 xlds[128 * 256];   // 64 KB : x tile, [pixel][c]
    __shared__ float  scal[128], biasl[128], wchs[128];
    __shared__ float  sps[128];
    __shared__ int    spx[128];

    const int n     = blockIdx.x >> 5;
    const int chunk = blockIdx.x & 31;
    const int t     = threadIdx.x;
    const int lane  = t & 31;
    const int wave  = t >> 5;
    const int Ml    = lane & 15;
    const int half  = lane >> 4;
    const int o_base = wave * 16;

    if (t < 128) {
        float s  = cg[t] * rsqrtf(cv[t] + EPSV);
        scal[t]  = s;
        biasl[t] = (conv1_b[t] - cm[t]) * s + cb[t];  // BN folded: g = relu(W'x + bias)
        wchs[t]  = ws[WS_WCH + n * CCH + t];
    }
    __syncthreads();
    for (int idx = t; idx < CCH * 256; idx += 256) {
        int o = idx >> 8;
        wlds[idx] = (__bf16)(conv1_w[idx] * scal[o]);
    }
    // (first per-tile __syncthreads covers wlds visibility)

    for (int tile = 0; tile < 4; ++tile) {
        const int pix0 = chunk * 512 + tile * 128;
        if (t < 128) { sps[t] = 0.f; spx[t] = 0; }

        // build modified f_cat tile in bf16: x[c]=f_ir*w+f_vi, x[c+128]=f_vi*w+f_ir
        {
            const int p     = t & 127;
            const int chalf = t >> 7;
            const size_t gbase = (size_t)n * CHW + (size_t)pix0 + p;
            for (int it = 0; it < 64; ++it) {
                int cc  = chalf + 2 * it;
                float a = f_vi[gbase + (size_t)cc * HWPX];
                float b = f_ir[gbase + (size_t)cc * HWPX];
                float w = wchs[cc];
                xlds[p * 256 + cc]       = (__bf16)fmaf(b, w, a);
                xlds[p * 256 + cc + 128] = (__bf16)fmaf(a, w, b);
            }
        }
        __syncthreads();

        v8f z = {0.f, 0.f, 0.f, 0.f, 0.f, 0.f, 0.f, 0.f};
        v8f acc[4];
#pragma unroll
        for (int j = 0; j < 4; ++j) acc[j] = z;

#pragma unroll
        for (int k = 0; k < 8; ++k) {
            const int k0 = k * 32;
            v16bf afr;  // A: 16x32 bf16 fragment per documented layout
#pragma unroll
            for (int i = 0; i < 8; ++i) {
                int kb = ((i < 4) ? (2 * i + 8 * half) : (16 + 2 * (i - 4) + 8 * half)) + k0;
                afr[2 * i]     = wlds[(o_base + Ml) * 256 + kb];
                afr[2 * i + 1] = wlds[(o_base + Ml) * 256 + kb + 1];
            }
#pragma unroll
            for (int j = 0; j < 4; ++j) {
                const int p = j * 16 + Ml;
                v16bf bfr;  // B: 32x16 bf16 fragment (K split across lane halves)
#pragma unroll
                for (int i = 0; i < 8; ++i) {
                    int kb = k0 + ((i < 4) ? (2 * i + 8 * half) : (16 + 2 * (i - 4) + 8 * half));
                    bfr[2 * i]     = xlds[p * 256 + kb];
                    bfr[2 * i + 1] = xlds[p * 256 + kb + 1];
                }
                acc[j] = __builtin_amdgcn_wmma_f32_16x16x32_bf16(
                    false, afr, false, bfr, (short)0, acc[j], false, false);
            }
        }

        // epilogue: bias+relu, per-channel sum/max, per-pixel sum/max
        float rowsum[8], rowmax[8];
#pragma unroll
        for (int r = 0; r < 8; ++r) { rowsum[r] = 0.f; rowmax[r] = 0.f; }
#pragma unroll
        for (int j = 0; j < 4; ++j) {
            const int p = j * 16 + Ml;
            float psum = 0.f, pmax = 0.f;
#pragma unroll
            for (int r = 0; r < 8; ++r) {
                int o = o_base + r + 8 * half;            // D layout: VGPR r -> M=r (+8 upper half)
                float v = fmaxf(acc[j][r] + biasl[o], 0.f);
                rowsum[r] += v;
                rowmax[r]  = fmaxf(rowmax[r], v);
                psum += v;
                pmax  = fmaxf(pmax, v);
            }
            atomicAdd(&sps[p], psum);
            atomicMax(&spx[p], __float_as_int(pmax));     // g >= 0
        }
#pragma unroll
        for (int r = 0; r < 8; ++r) {
            float v = rowsum[r];
            float m = rowmax[r];
#pragma unroll
            for (int msk = 1; msk < 16; msk <<= 1) {
                v += __shfl_xor(v, msk, 32);
                m  = fmaxf(m, __shfl_xor(m, msk, 32));
            }
            if ((lane & 15) == 0) {
                int o = o_base + r + 8 * half;
                atomicAdd(ws + WS_AVGS + n * CCH + o, v);
                atomicMax((int*)ws + WS_GMAX + n * CCH + o, __float_as_int(m));
            }
        }
        __syncthreads();
        if (t < 128) {
            ws[WS_SPMEAN + n * HWPX + pix0 + t] = sps[t] * (1.f / CCH);
            ws[WS_SPMAX  + n * HWPX + pix0 + t] = __int_as_float(spx[t]);
        }
        __syncthreads();
    }
}

// ---------------- CBAM channel MLP -> ca ----------------
__global__ void k_attn2(float* ws, const float* w1, const float* w2) {
    __shared__ float t1a[16 * 8], t1m[16 * 8];
    int t = threadIdx.x;
    if (t < 128) {
        int n = t >> 3, r = t & 7;
        float aa = 0.f, mm = 0.f;
        const float* wr = w1 + r * 128;
        for (int c = 0; c < 128; ++c) {
            aa += (ws[WS_AVGS + n * 128 + c] * (1.f / HWPX)) * wr[c];
            mm += ws[WS_GMAX + n * 128 + c] * wr[c];
        }
        t1a[t] = fmaxf(aa, 0.f);
        t1m[t] = fmaxf(mm, 0.f);
    }
    __syncthreads();
    for (int idx = t; idx < 16 * 128; idx += 256) {
        int n = idx >> 7, o = idx & 127;
        float s = 0.f;
        for (int r = 0; r < 8; ++r) s += (t1a[n * 8 + r] + t1m[n * 8 + r]) * w2[o * 8 + r];
        ws[WS_CA + idx] = sigmoidf(s);
    }
}

// ---------------- 7x7 spatial conv + sigmoid -> sa ----------------
__global__ void k_saconv(float* ws, const float* __restrict__ sa_w) {
    __shared__ float wk[98];
    int t = threadIdx.x;
    if (t < 98) wk[t] = sa_w[t];
    __syncthreads();
    int id  = blockIdx.x * 256 + t;     // grid sized exactly NB*HWPX
    int n   = id >> 14;
    int pix = id & 16383;
    int h = pix >> 7, w = pix & 127;
    const float* spm = ws + WS_SPMEAN + n * HWPX;
    const float* spx = ws + WS_SPMAX  + n * HWPX;
    float acc = 0.f;
    for (int ky = 0; ky < 7; ++ky) {
        int hh = h + ky - 3;
        if (hh < 0 || hh >= 128) continue;
        for (int kx = 0; kx < 7; ++kx) {
            int wp = w + kx - 3;
            if (wp < 0 || wp >= 128) continue;
            int q = hh * 128 + wp;
            acc += wk[ky * 7 + kx] * spm[q] + wk[49 + ky * 7 + kx] * spx[q];
        }
    }
    ws[WS_SA + id] = sigmoidf(acc);
}

// ---------------- final blend (float4 streaming) ----------------
__global__ void __launch_bounds__(256)
k_final(const float* __restrict__ f_vi, const float* __restrict__ f_ir,
        const float* __restrict__ ws, float* __restrict__ out) {
    size_t e   = ((size_t)blockIdx.x * 256 + threadIdx.x) * 4;
    int    n   = (int)(e / ((size_t)CCH * HWPX));
    size_t rem = e % ((size_t)CCH * HWPX);
    int    c   = (int)(rem / HWPX);
    int    pix = (int)(rem % HWPX);
    float wc  = ws[WS_WCH + n * CCH + c];
    float cac = ws[WS_CA  + n * CCH + c];
    float4 a4 = *(const float4*)(f_vi + e);
    float4 b4 = *(const float4*)(f_ir + e);
    float4 s4 = *(const float4*)(ws + WS_SA + n * HWPX + pix);
    float4 o4;
    {
        float wgt = sigmoidf(s4.x * cac);
        o4.x = wgt * fmaf(b4.x, wc, a4.x) + (1.f - wgt) * fmaf(a4.x, wc, b4.x);
    }
    {
        float wgt = sigmoidf(s4.y * cac);
        o4.y = wgt * fmaf(b4.y, wc, a4.y) + (1.f - wgt) * fmaf(a4.y, wc, b4.y);
    }
    {
        float wgt = sigmoidf(s4.z * cac);
        o4.z = wgt * fmaf(b4.z, wc, a4.z) + (1.f - wgt) * fmaf(a4.z, wc, b4.z);
    }
    {
        float wgt = sigmoidf(s4.w * cac);
        o4.w = wgt * fmaf(b4.w, wc, a4.w) + (1.f - wgt) * fmaf(a4.w, wc, b4.w);
    }
    *(float4*)(out + e) = o4;
}

extern "C" void kernel_launch(void* const* d_in, const int* in_sizes, int n_in,
                              void* d_out, int out_size, void* d_ws, size_t ws_size,
                              hipStream_t stream) {
    (void)in_sizes; (void)n_in; (void)out_size; (void)ws_size;
    const float* f_vi    = (const float*)d_in[0];
    const float* f_ir    = (const float*)d_in[1];
    const float* ca1_w   = (const float*)d_in[2];
    const float* ca1_b   = (const float*)d_in[3];
    const float* bn_a_g  = (const float*)d_in[4];
    const float* bn_a_b  = (const float*)d_in[5];
    const float* bn_a_m  = (const float*)d_in[6];
    const float* bn_a_v  = (const float*)d_in[7];
    const float* ca2_w   = (const float*)d_in[8];
    const float* ca2_b   = (const float*)d_in[9];
    const float* bn_b_g  = (const float*)d_in[10];
    const float* bn_b_b  = (const float*)d_in[11];
    const float* bn_b_m  = (const float*)d_in[12];
    const float* bn_b_v  = (const float*)d_in[13];
    const float* conv1_w = (const float*)d_in[14];
    const float* conv1_b = (const float*)d_in[15];
    const float* bn_c_g  = (const float*)d_in[16];
    const float* bn_c_b  = (const float*)d_in[17];
    const float* bn_c_m  = (const float*)d_in[18];
    const float* bn_c_v  = (const float*)d_in[19];
    const float* chatt_w1= (const float*)d_in[20];
    const float* chatt_w2= (const float*)d_in[21];
    const float* sa_w    = (const float*)d_in[22];
    float* out = (float*)d_out;
    float* ws  = (float*)d_ws;

    k_init  <<<16, 256, 0, stream>>>(ws);
    k_pool  <<<NB * 256, 256, 0, stream>>>(f_vi, f_ir, ws);
    k_attn1 <<<1, 256, 0, stream>>>(ws, ca1_w, ca1_b, bn_a_g, bn_a_b, bn_a_m, bn_a_v,
                                    ca2_w, ca2_b, bn_b_g, bn_b_b, bn_b_m, bn_b_v);
    k_main  <<<NB * 32, 256, 0, stream>>>(f_vi, f_ir, conv1_w, conv1_b,
                                          bn_c_g, bn_c_b, bn_c_m, bn_c_v, ws);
    k_attn2 <<<1, 256, 0, stream>>>(ws, chatt_w1, chatt_w2);
    k_saconv<<<(NB * HWPX) / 256, 256, 0, stream>>>(ws, sa_w);
    k_final <<<(NB * CHW) / (256 * 4), 256, 0, stream>>>(f_vi, f_ir, ws, out);
}
